// TreeLstm_85109071937882
// MI455X (gfx1250) — matrix operand order repaired and checked by reading, not compiled, not run on hardware
//
#include <hip/hip_runtime.h>

// Tree-LSTM over L=16384 tokens, D=256, 5 gates (5D=1280), NL=5 proj.
// bf16 WMMA (f32 accum) GEMMs; weight B-tiles staged to LDS by the Tensor
// Data Mover (tensor_load_to_lds, TENSORcnt) in the hot per-level kernel;
// one kernel launch per tree level (stream-chained, graph-safe).

#define D_    256
#define G5_   1280
#define L_    16384
#define NL_   5
#define RT_   2                      // row tiles (16 rows each) per wave
#define WAVES_ 8                     // waves per workgroup
#define WGROWS_ (WAVES_ * RT_ * 16)  // 256 output rows per workgroup

typedef __attribute__((ext_vector_type(16))) __bf16 v16bf;
typedef __attribute__((ext_vector_type(8)))  __bf16 v8bf;
typedef __attribute__((ext_vector_type(8)))  float  v8f;
typedef __attribute__((ext_vector_type(4)))  unsigned int v4u;
typedef __attribute__((ext_vector_type(8)))  unsigned int v8u;

union Frag {
    v16bf v;
    v8bf  h[2];
};

__device__ __forceinline__ float sigf(float x) {
    return 1.0f / (1.0f + __expf(-x));
}
__device__ __forceinline__ float tanh_f(float x) {
    float e = __expf(-2.0f * x);
    return (1.0f - e) / (1.0f + e);
}

// A fragment (16x32 bf16) from a row-major bf16 matrix, given this lane's row
// pointer. ISA 16-bit A layout: lanes 0-15 hold K=k0+0..7 and k0+16..23 of
// row M=lane; lanes 16-31 hold K=k0+8..15 and k0+24..31.
__device__ __forceinline__ Frag load_a_frag(const __bf16* rowPtr, int k0, int lane) {
    const int kh = (lane < 16) ? 0 : 8;
    Frag f;
    f.h[0] = *(const v8bf*)(rowPtr + k0 + kh);
    f.h[1] = *(const v8bf*)(rowPtr + k0 + kh + 16);
    return f;
}

// B fragment (32x16 bf16) from LDS-staged tiles laid out [tile][n][k] with
// k contiguous (256 per (tile,n) row). B layout (matches A): lanes 0-15 hold
// K=k0+0..7 and k0+16..23 of column n=lane; lanes 16-31 hold K=k0+8..15 and k0+24..31.
__device__ __forceinline__ Frag load_b_lds(const __bf16* ldsBase, int tile,
                                           int k0, int lane) {
    const int n  = lane & 15;
    const int kb = k0 + ((lane < 16) ? 0 : 8);
    const __bf16* p = ldsBase + ((tile * 16 + n) * D_ + kb);
    Frag f;
    f.h[0] = *(const v8bf*)(p);
    f.h[1] = *(const v8bf*)(p + 16);
    return f;
}

__device__ __forceinline__ v8f wmma_bf16(const Frag& a, const Frag& b, v8f c) {
    return __builtin_amdgcn_wmma_f32_16x16x32_bf16(
        /*neg_a=*/false, a.v, /*neg_b=*/false, b.v,
        /*c_mod=*/(short)0, c, /*reuse_a=*/false, /*reuse_b=*/false);
}

// ---------------------------------------------------------------------------
// TDM: DMA one 3D weight tile-set (5 gates x 16 cols x 256 k, bf16) from
// global (transposed-weight layout WT[row=g*256+col][k], k contiguous) into
// LDS at ldsByteAddr.  D# per CDNA5 ISA:
//   dim0 = k   : tile 256, tensor 256, stride 256 elems (contiguous rows)
//   dim1 = col : tile 16,  tensor 16,  stride 65536 elems between gates
//   dim2 = gate: tile 5,   tensor 5
// data_size=1 (2 bytes).  Tracked with TENSORcnt.
// ---------------------------------------------------------------------------
__device__ __forceinline__ void tdm_load_gate_tiles(const __bf16* gptr,
                                                    unsigned int ldsByteAddr) {
    unsigned long long ga = (unsigned long long)(size_t)gptr;
    v4u g0;
    g0[0] = 1u;                                   // count=1, user descriptor
    g0[1] = ldsByteAddr;                          // lds_addr[31:0]
    g0[2] = (unsigned int)(ga & 0xFFFFFFFFu);     // global_addr[31:0]
    g0[3] = ((unsigned int)(ga >> 32) & 0x01FFFFFFu) | 0x80000000u; // type=2
    v8u g1;
    g1[0] = 0x00010000u;   // workgroup_mask=0 (no cluster), data_size=1 (2B)
    g1[1] = (256u << 16);  // atomic_barrier_addr=0; tensor_dim0[15:0]=256
    g1[2] = (16u << 16);   // tensor_dim0[31:16]=0; tensor_dim1[15:0]=16
    g1[3] = (256u << 16);  // tensor_dim1[31:16]=0; tile_dim0=256
    g1[4] = (5u << 16) | 16u;  // tile_dim1=16, tile_dim2=5
    g1[5] = 256u;          // tensor_dim0_stride[31:0] = 256 elems
    g1[6] = 0u;            // dim0_stride[47:32]=0; dim1_stride[15:0]=0
    g1[7] = 1u;            // tensor_dim1_stride[47:16] = 1 -> 65536 elems
    v4u g2;
    g2[0] = 5u;            // tensor_dim2 = 5
    g2[1] = 0u; g2[2] = 0u; g2[3] = 0u;
    v4u g3;
    g3[0] = 0u; g3[1] = 0u; g3[2] = 0u; g3[3] = 0u;
    asm volatile("tensor_load_to_lds %0, %1, %2, %3"
                 :
                 : "s"(g0), "s"(g1), "s"(g2), "s"(g3)
                 : "memory");
}

// ---------------------------------------------------------------------------
// One-time: Wx, Ul, Ur (fp32 [256 x 1280] row-major) -> transposed bf16
// WT[n*256 + k]  (column-of-output n contiguous in reduction k).
// ---------------------------------------------------------------------------
__global__ void prep_weights(const float* __restrict__ Wx,
                             const float* __restrict__ Ul,
                             const float* __restrict__ Ur,
                             __bf16* __restrict__ WxT,
                             __bf16* __restrict__ UlT,
                             __bf16* __restrict__ UrT) {
    int idx = blockIdx.x * blockDim.x + threadIdx.x;
    if (idx >= G5_ * D_) return;
    int n = idx / D_;
    int k = idx - n * D_;
    size_t src = (size_t)k * G5_ + n;
    WxT[idx] = (__bf16)Wx[src];
    UlT[idx] = (__bf16)Ul[src];
    UrT[idx] = (__bf16)Ur[src];
}

// ---------------------------------------------------------------------------
// Gather + convert: xbf[i][k] = bf16(emb[tokens[i]][k]).  Streaming, ~25 MB.
// ---------------------------------------------------------------------------
__global__ void gather_embed(const int* __restrict__ tokens,
                             const float* __restrict__ emb,
                             __bf16* __restrict__ xbf) {
    int idx = blockIdx.x * blockDim.x + threadIdx.x;
    if (idx >= L_ * D_) return;
    int row = idx >> 8;
    int k   = idx & (D_ - 1);
    xbf[idx] = (__bf16)emb[(size_t)tokens[row] * D_ + k];
}

// ---------------------------------------------------------------------------
// Leaf level: pre = x @ Wx + b over gates {i,o,u} (c_l=c_r=0);
// c = sig(i)*tanh(u); h = sig(o)*tanh(c).
// Workgroup: 8 waves, one 16-col block, 256 rows; B tiles staged in LDS
// with vector loads (3 non-adjacent gate tiles).  Dynamic LDS: 24 KB.
// ---------------------------------------------------------------------------
__global__ void __launch_bounds__(256)
leaf_kernel(const __bf16* __restrict__ xbf, const __bf16* __restrict__ WxT,
            const float* __restrict__ bias, float* __restrict__ hiddenL0,
            __bf16* __restrict__ hbf, float* __restrict__ cbuf) {
    extern __shared__ __bf16 ldsB[];          // [3][16][256]
    const int tid  = threadIdx.x;
    const int wave = tid >> 5;
    const int lane = tid & 31;
    const int colg   = blockIdx.x * 16;
    const int wgbase = blockIdx.y * WGROWS_;
    const int GATES[3] = {0, 3, 4};           // i, o, u

    // Stage B tiles (coalesced along k).
    for (int c = tid; c < 3 * 16 * (D_ / 8); c += 256) {
        int row = c >> 5;                     // 32 v8bf chunks per (tile,n)
        int j   = (c & 31) * 8;
        int gi  = row >> 4;
        int n   = row & 15;
        const __bf16* src = WxT + ((size_t)(GATES[gi] * D_ + colg + n) * D_ + j);
        *(v8bf*)&ldsB[row * D_ + j] = *(const v8bf*)src;
    }
    __syncthreads();

    const int wbase = wgbase + wave * (RT_ * 16);
    const int m = lane & 15;
    const __bf16* arow[RT_];
#pragma unroll
    for (int t = 0; t < RT_; ++t)
        arow[t] = xbf + (size_t)(wbase + t * 16 + m) * D_;

    v8f acc[3][RT_] = {};
    for (int k0 = 0; k0 < D_; k0 += 32) {
        Frag a[RT_];
#pragma unroll
        for (int t = 0; t < RT_; ++t) a[t] = load_a_frag(arow[t], k0, lane);
#pragma unroll
        for (int gi = 0; gi < 3; ++gi) {
            Frag bf = load_b_lds(ldsB, gi, k0, lane);
#pragma unroll
            for (int t = 0; t < RT_; ++t)
                acc[gi][t] = wmma_bf16(a[t], bf, acc[gi][t]);
        }
    }

    const int n   = lane & 15;
    const int col = colg + n;
    const float bi = bias[0 * D_ + col];
    const float bo = bias[3 * D_ + col];
    const float bu = bias[4 * D_ + col];
    const int mbase = (lane < 16) ? 0 : 8;
#pragma unroll
    for (int t = 0; t < RT_; ++t) {
#pragma unroll
        for (int r = 0; r < 8; ++r) {
            int row = wbase + t * 16 + mbase + r;
            float iv = sigf(acc[0][t][r] + bi);
            float ov = sigf(acc[1][t][r] + bo);
            float uv = tanh_f(acc[2][t][r] + bu);
            float c  = iv * uv;
            float h  = ov * tanh_f(c);
            size_t off = (size_t)row * D_ + col;
            hiddenL0[off] = h;
            hbf[off]      = (__bf16)h;
            cbuf[off]     = c;
        }
    }
}

// ---------------------------------------------------------------------------
// Tree level: pre = hl @ Ul + hr @ Ur + b, full LSTM cell with cl, cr.
// hl = hprev[0::2], hr = hprev[1::2]; N rows out, Nprev = 2N rows in.
// B tiles (10 x 16 x 256 bf16 = 80 KB dynamic LDS; tiles 0-4 = Ul gates,
// 5-9 = Ur gates) are DMA'd by the Tensor Data Mover: wave 0 issues two
// tensor_load_to_lds ops, waits TENSORcnt==0, then the workgroup barrier
// publishes the LDS contents to all 8 waves.
// ---------------------------------------------------------------------------
__global__ void __launch_bounds__(256)
level_kernel(const __bf16* __restrict__ hprev, const float* __restrict__ cprev,
             const __bf16* __restrict__ UlT, const __bf16* __restrict__ UrT,
             const float* __restrict__ bias,
             float* __restrict__ hiddenOut, __bf16* __restrict__ hbfOut,
             float* __restrict__ cOut, int N, int Nprev) {
    extern __shared__ __bf16 ldsB[];          // [10][16][256]
    const int tid  = threadIdx.x;
    const int wave = tid >> 5;
    const int lane = tid & 31;
    const int colg   = blockIdx.x * 16;
    const int wgbase = blockIdx.y * WGROWS_;

    if (wave == 0) {
        // Low 32 bits of the flat shared address = LDS byte offset.
        unsigned int ldsBase = (unsigned int)(size_t)(&ldsB[0]);
        tdm_load_gate_tiles(UlT + (size_t)colg * D_, ldsBase);
        tdm_load_gate_tiles(UrT + (size_t)colg * D_, ldsBase + 5 * 16 * D_ * 2);
        __builtin_amdgcn_s_wait_tensorcnt(0);
    }
    __syncthreads();

    const int wbase = wgbase + wave * (RT_ * 16);
    if (wbase >= N) return;                   // wave-uniform skip (post-barrier)

    const int m = lane & 15;
    const __bf16* arl[RT_];
    const __bf16* arr[RT_];
#pragma unroll
    for (int t = 0; t < RT_; ++t) {
        int rowm = wbase + t * 16 + m;
        int rl = 2 * rowm;     if (rl > Nprev - 1) rl = Nprev - 1;
        int rr = 2 * rowm + 1; if (rr > Nprev - 1) rr = Nprev - 1;
        arl[t] = hprev + (size_t)rl * D_;
        arr[t] = hprev + (size_t)rr * D_;
    }

    v8f acc[5][RT_] = {};
    for (int k0 = 0; k0 < D_; k0 += 32) {
        Frag al[RT_], ar[RT_];
#pragma unroll
        for (int t = 0; t < RT_; ++t) {
            al[t] = load_a_frag(arl[t], k0, lane);
            ar[t] = load_a_frag(arr[t], k0, lane);
        }
#pragma unroll
        for (int g = 0; g < 5; ++g) {
            Frag bl = load_b_lds(ldsB, g, k0, lane);
#pragma unroll
            for (int t = 0; t < RT_; ++t)
                acc[g][t] = wmma_bf16(al[t], bl, acc[g][t]);
            Frag br = load_b_lds(ldsB, 5 + g, k0, lane);
#pragma unroll
            for (int t = 0; t < RT_; ++t)
                acc[g][t] = wmma_bf16(ar[t], br, acc[g][t]);
        }
    }

    const int n   = lane & 15;
    const int col = colg + n;
    const float bi  = bias[0 * D_ + col];
    const float bfl = bias[1 * D_ + col];
    const float bfr = bias[2 * D_ + col];
    const float bo  = bias[3 * D_ + col];
    const float bu  = bias[4 * D_ + col];
    const int mbase = (lane < 16) ? 0 : 8;
#pragma unroll
    for (int t = 0; t < RT_; ++t) {
#pragma unroll
        for (int r = 0; r < 8; ++r) {
            int row = wbase + t * 16 + mbase + r;
            if (row < N) {
                float cl = cprev[(size_t)(2 * row) * D_ + col];
                float cr = cprev[(size_t)(2 * row + 1) * D_ + col];
                float iv  = sigf(acc[0][t][r] + bi);
                float flv = sigf(acc[1][t][r] + bfl);
                float frv = sigf(acc[2][t][r] + bfr);
                float ov  = sigf(acc[3][t][r] + bo);
                float uv  = tanh_f(acc[4][t][r] + bu);
                float c = iv * uv + flv * cl + frv * cr;
                float h = ov * tanh_f(c);
                size_t off = (size_t)row * D_ + col;
                hiddenOut[off] = h;
                hbfOut[off]    = (__bf16)h;
                cOut[off]      = c;
            }
        }
    }
}

// ---------------------------------------------------------------------------
// Final projection: out = hidden @ Wp + bp   (32767 x 256) @ (256 x 5)
// ---------------------------------------------------------------------------
__global__ void proj_kernel(const float* __restrict__ hidden,
                            const float* __restrict__ Wp,
                            const float* __restrict__ bp,
                            float* __restrict__ out, int rows) {
    int idx = blockIdx.x * blockDim.x + threadIdx.x;
    if (idx >= rows * NL_) return;
    int row = idx / NL_;
    int p   = idx - row * NL_;
    const float* hr = hidden + (size_t)row * D_;
    float s = bp[p];
#pragma unroll 4
    for (int k = 0; k < D_; ++k) s += hr[k] * Wp[k * NL_ + p];
    out[idx] = s;
}

// ---------------------------------------------------------------------------
extern "C" void kernel_launch(void* const* d_in, const int* in_sizes, int n_in,
                              void* d_out, int out_size, void* d_ws, size_t ws_size,
                              hipStream_t stream) {
    const int*   tokens = (const int*)d_in[0];
    const float* emb    = (const float*)d_in[1];
    const float* Wx     = (const float*)d_in[2];
    const float* Ul     = (const float*)d_in[3];
    const float* Ur     = (const float*)d_in[4];
    const float* b      = (const float*)d_in[5];
    const float* Wp     = (const float*)d_in[6];
    const float* bp     = (const float*)d_in[7];
    float*       out    = (float*)d_out;

    // Workspace carve-out (~95 MB total).
    char* ws = (char*)d_ws;
    size_t off = 0;
    auto carve = [&](size_t bytes) -> void* {
        void* p = ws + off;
        off += (bytes + 255) & ~(size_t)255;
        return p;
    };
    __bf16* WxT  = (__bf16*)carve((size_t)G5_ * D_ * 2);
    __bf16* UlT  = (__bf16*)carve((size_t)G5_ * D_ * 2);
    __bf16* UrT  = (__bf16*)carve((size_t)G5_ * D_ * 2);
    __bf16* xbf  = (__bf16*)carve((size_t)L_ * D_ * 2);
    __bf16* hbfA = (__bf16*)carve((size_t)L_ * D_ * 2);
    __bf16* hbfB = (__bf16*)carve((size_t)L_ * D_ * 2);
    float*  cA   = (float*)carve((size_t)L_ * D_ * 4);
    float*  cB   = (float*)carve((size_t)L_ * D_ * 4);
    float*  hidden = (float*)carve((size_t)(2 * L_ - 1) * D_ * 4);

    // 1. Weight convert + transpose to bf16.
    {
        int elems = G5_ * D_;
        prep_weights<<<dim3((elems + 255) / 256), dim3(256), 0, stream>>>(
            Wx, Ul, Ur, WxT, UlT, UrT);
    }

    // 2. Embedding gather -> bf16.
    {
        int elems = L_ * D_;
        gather_embed<<<dim3((elems + 255) / 256), dim3(256), 0, stream>>>(
            tokens, emb, xbf);
    }

    // 3. Leaf level (16384 rows). 24 KB dynamic LDS.
    leaf_kernel<<<dim3(16, L_ / WGROWS_), dim3(256), 3 * 16 * D_ * 2, stream>>>(
        xbf, WxT, b, hidden, hbfA, cA);

    // 4. Tree levels 8192 -> 1, chained on the stream. 80 KB dynamic LDS.
    size_t hoff = L_;
    __bf16* hp = hbfA; __bf16* hn = hbfB;
    float*  cp = cA;   float*  cn = cB;
    int Nprev = L_;
    for (int N = L_ / 2; N >= 1; N >>= 1) {
        int rb = (N + WGROWS_ - 1) / WGROWS_;
        level_kernel<<<dim3(16, rb), dim3(256), 10 * 16 * D_ * 2, stream>>>(
            hp, cp, UlT, UrT, b, hidden + hoff * D_, hn, cn, N, Nprev);
        hoff += N;
        { __bf16* t = hp; hp = hn; hn = t; }
        { float*  t = cp; cp = cn; cn = t; }
        Nprev = N;
    }

    // 5. Projection over all 32767 concatenated hidden rows.
    {
        int rows = 2 * L_ - 1;
        proj_kernel<<<dim3((rows * NL_ + 255) / 256), dim3(256), 0, stream>>>(
            hidden, Wp, bp, out, rows);
    }
}
